// DE_TGraph_24240795419241
// MI455X (gfx1250) — matrix operand order repaired and checked by reading, not compiled, not run on hardware
//
#include <hip/hip_runtime.h>
#include <hip/hip_bf16.h>

typedef __attribute__((ext_vector_type(16))) _Float16 v16h;
typedef __attribute__((ext_vector_type(8)))  float    v8f;

#define NN 32768   // neighbors
#define QQ 4096    // query entities
#define EE 32768   // pooling edges
#define BB 1024    // batch triples
#define R2 460     // 2 * NUM_REL
#define DIM 128
#define SDIM 96
#define TDIM 32

// ---------------------------------------------------------------- features
__global__ __launch_bounds__(128)
void compute_x_kernel(const int* __restrict__ nidx,
                      const float* __restrict__ years,
                      const float* __restrict__ months,
                      const float* __restrict__ days,
                      const float* __restrict__ ent,
                      const float* __restrict__ yf, const float* __restrict__ yp, const float* __restrict__ ya,
                      const float* __restrict__ mf, const float* __restrict__ mp, const float* __restrict__ ma,
                      const float* __restrict__ df, const float* __restrict__ dp, const float* __restrict__ da,
                      _Float16* __restrict__ xh) {
  const int n = blockIdx.x;
  const int d = threadIdx.x;
  const int e = nidx[n];
  float v;
  if (d < SDIM) {
    v = ent[(size_t)e * SDIM + d];
  } else {
    const int j = d - SDIM;
    const size_t o = (size_t)e * TDIM + j;
    const float ty = ya[o] * sinf(yf[o] * years[n]  + yp[o]);
    const float tm = ma[o] * sinf(mf[o] * months[n] + mp[o]);
    const float td = da[o] * sinf(df[o] * days[n]   + dp[o]);
    v = ty + tm + td;
  }
  xh[(size_t)n * DIM + d] = (_Float16)v;
}

// ---------------------------------------------------------------- W -> f16
__global__ void convert_w_kernel(const float* __restrict__ W, _Float16* __restrict__ Wh, int total) {
  for (int i = blockIdx.x * blockDim.x + threadIdx.x; i < total; i += gridDim.x * blockDim.x)
    Wh[i] = (_Float16)W[i];
}

// ---------------------------------------------------------------- bucketing
__global__ void hist_kernel(const int* __restrict__ key, int* __restrict__ cnt, int n) {
  int i = blockIdx.x * blockDim.x + threadIdx.x;
  if (i < n) atomicAdd(&cnt[key[i]], 1);
}

__global__ void exscan_kernel(const int* __restrict__ cnt, int* __restrict__ off, int n) {
  if (blockIdx.x == 0 && threadIdx.x == 0) {
    int acc = 0;
    for (int i = 0; i < n; ++i) { off[i] = acc; acc += cnt[i]; }
    off[n] = acc;
  }
}

__global__ void scatter_kernel(const int* __restrict__ key, const int* __restrict__ off,
                               int* __restrict__ cur, int* __restrict__ perm, int n) {
  int i = blockIdx.x * blockDim.x + threadIdx.x;
  if (i < n) {
    int k = key[i];
    int pos = off[k] + atomicAdd(&cur[k], 1);
    perm[pos] = i;
  }
}

// ---------------------------------------------------------------- WMMA GEMM
// One block per relation (460 blocks, 256 thr = 8 wave32).
// Wave w owns output columns [16w, 16w+16): keeps B = W[r] f16 fragments for
// all 4 K-chunks resident, sweeps 16-row A tiles (gathered via perm) through
// v_wmma_f32_16x16x32_f16 with f32 accumulation, adds bias, scatters z rows.
__global__ __launch_bounds__(256)
void gemm_per_rel_kernel(const _Float16* __restrict__ xh,
                         const _Float16* __restrict__ Wh,
                         const float* __restrict__ bvec,
                         const int* __restrict__ off,
                         const int* __restrict__ perm,
                         float* __restrict__ z) {
  const int r = blockIdx.x;
  const int o0 = off[r];
  const int cnt = off[r + 1] - o0;
  if (cnt == 0) return;

  const int wave = threadIdx.x >> 5;     // 0..7 -> output col tile
  const int lane = threadIdx.x & 31;
  const int m = lane & 15;
  const int h = lane >> 4;
  const int colBase = wave * 16;
  const _Float16* __restrict__ Wr = Wh + (size_t)r * DIM * DIM;

  // B fragments: B[k][n], n = colBase + m, K half selected by lane half.
  v16h bfrag[4];
#pragma unroll
  for (int c = 0; c < 4; ++c) {
#pragma unroll
    for (int j = 0; j < 8; ++j) {
      const int k0 = c * 32 + h * 16 + 2 * j;
      bfrag[c][2 * j]     = Wr[(size_t)k0       * DIM + colBase + m];
      bfrag[c][2 * j + 1] = Wr[(size_t)(k0 + 1) * DIM + colBase + m];
    }
  }

  const int ntiles = (cnt + 15) >> 4;
  for (int t = 0; t < ntiles; ++t) {
    const int rowIdx = t * 16 + m;
    const int row = perm[o0 + (rowIdx < cnt ? rowIdx : 0)];
    const _Float16* __restrict__ xr = xh + (size_t)row * DIM;

    v8f acc = {};
#pragma unroll
    for (int c = 0; c < 4; ++c) {
      v16h afrag;
#pragma unroll
      for (int j = 0; j < 4; ++j) {
        const int k0 = c * 32 + h * 16 + 2 * j;       // VGPR 0..3
        afrag[2 * j]         = xr[k0];
        afrag[2 * j + 1]     = xr[k0 + 1];
        const int k1 = k0 + 8;                         // VGPR 4..7
        afrag[8 + 2 * j]     = xr[k1];
        afrag[8 + 2 * j + 1] = xr[k1 + 1];
      }
      acc = __builtin_amdgcn_wmma_f32_16x16x32_f16(
          false, afrag, false, bfrag[c], (short)0, acc, false, false);
    }

    // C/D layout: VGPR v -> M = v + 8*h, N = colBase + m
#pragma unroll
    for (int v = 0; v < 8; ++v) {
      const int M = v + 8 * h;
      const int ri = t * 16 + M;
      if (ri < cnt) {
        const int grow = perm[o0 + ri];
        const int col = colBase + m;
        z[(size_t)grow * DIM + col] = acc[v] + bvec[(size_t)r * DIM + col];
      }
    }
  }
}

// ---------------------------------------------------------------- batchnorm
__global__ __launch_bounds__(128)
void bn_stats_kernel(const float* __restrict__ z, const int* __restrict__ off,
                     const int* __restrict__ perm,
                     float* __restrict__ mean, float* __restrict__ invstd) {
  const int r = blockIdx.x;
  const int d = threadIdx.x;
  const int o0 = off[r];
  const int cnt = off[r + 1] - o0;
  float s = 0.f, s2 = 0.f;
  for (int i = 0; i < cnt; ++i) {
    const float v = z[(size_t)perm[o0 + i] * DIM + d];
    s += v; s2 += v * v;
  }
  const float den = (float)(cnt > 1 ? cnt : 1);
  const float mu = s / den;
  const float var = fmaxf(s2 / den - mu * mu, 0.f);
  mean[(size_t)r * DIM + d]   = mu;
  invstd[(size_t)r * DIM + d] = rsqrtf(var + 1e-5f);
}

__global__ __launch_bounds__(128)
void bn_relu_kernel(float* __restrict__ z, const int* __restrict__ rel_id,
                    const int* __restrict__ off,
                    const float* __restrict__ mean, const float* __restrict__ invstd,
                    const float* __restrict__ gamma, const float* __restrict__ beta) {
  const int n = blockIdx.x;
  const int d = threadIdx.x;
  const int r = rel_id[n];
  const int cnt = off[r + 1] - off[r];
  const size_t zi = (size_t)n * DIM + d;
  const size_t ri = (size_t)r * DIM + d;
  float v = z[zi];
  if (cnt > 1) v = (v - mean[ri]) * invstd[ri] * gamma[ri] + beta[ri];
  z[zi] = fmaxf(v, 0.f);
}

// ---------------------------------------------------------------- pooling
__global__ __launch_bounds__(128)
void pool_avg_kernel(const float* __restrict__ z, const int* __restrict__ psrc,
                     const int* __restrict__ poff, const int* __restrict__ pperm,
                     float* __restrict__ emb) {
  const int q = blockIdx.x;
  const int d = threadIdx.x;
  const int o0 = poff[q];
  const int cnt = poff[q + 1] - o0;
  float s = 0.f;
  for (int i = 0; i < cnt; ++i) {
    const int e = pperm[o0 + i];
    s += z[(size_t)psrc[e] * DIM + d];
  }
  emb[(size_t)q * DIM + d] = s / (float)(cnt > 0 ? cnt : 1);
}

// ---------------------------------------------------------------- scoring
__global__ __launch_bounds__(128)
void score_kernel(const float* __restrict__ emb, const float* __restrict__ rel_embs,
                  const int* __restrict__ head, const int* __restrict__ tail,
                  const int* __restrict__ rels, float* __restrict__ out) {
  const int i = blockIdx.x;
  const int d = threadIdx.x;
  const float diff = emb[(size_t)head[i] * DIM + d]
                   + rel_embs[(size_t)rels[i] * DIM + d]
                   - emb[(size_t)tail[i] * DIM + d];
  float s = diff * diff;
#pragma unroll
  for (int o = 16; o > 0; o >>= 1) s += __shfl_down(s, o, 32);
  __shared__ float red[4];
  if ((d & 31) == 0) red[d >> 5] = s;
  __syncthreads();
  if (d == 0) out[i] = -sqrtf(red[0] + red[1] + red[2] + red[3]);
}

// ---------------------------------------------------------------- launch
extern "C" void kernel_launch(void* const* d_in, const int* in_sizes, int n_in,
                              void* d_out, int out_size, void* d_ws, size_t ws_size,
                              hipStream_t stream) {
  const int*   neighbor_idx = (const int*)d_in[0];
  const int*   rel_id       = (const int*)d_in[1];
  const float* years        = (const float*)d_in[2];
  const float* months       = (const float*)d_in[3];
  const float* days         = (const float*)d_in[4];
  const int*   pool_src     = (const int*)d_in[5];
  const int*   pool_dst     = (const int*)d_in[6];
  const int*   head_pos     = (const int*)d_in[7];
  const int*   tail_pos     = (const int*)d_in[8];
  const int*   rels         = (const int*)d_in[9];
  const float* ent_embs     = (const float*)d_in[10];
  const float* y_freq = (const float*)d_in[11];
  const float* y_phi  = (const float*)d_in[12];
  const float* y_amp  = (const float*)d_in[13];
  const float* m_freq = (const float*)d_in[14];
  const float* m_phi  = (const float*)d_in[15];
  const float* m_amp  = (const float*)d_in[16];
  const float* d_freq = (const float*)d_in[17];
  const float* d_phi  = (const float*)d_in[18];
  const float* d_amp  = (const float*)d_in[19];
  const float* rel_embs = (const float*)d_in[20];
  const float* W        = (const float*)d_in[21];
  const float* bvec     = (const float*)d_in[22];
  const float* gamma    = (const float*)d_in[23];
  const float* beta     = (const float*)d_in[24];
  float* out = (float*)d_out;

  // Workspace carve-up (~43 MB total).
  char* base = (char*)d_ws;
  size_t o = 0;
  auto take = [&](size_t bytes) -> void* {
    void* p = base + o;
    o += (bytes + 255) & ~(size_t)255;
    return p;
  };
  _Float16* xh     = (_Float16*)take((size_t)NN * DIM * 2);        // 8 MB
  _Float16* Wh     = (_Float16*)take((size_t)R2 * DIM * DIM * 2);  // 15 MB
  float*    z      = (float*)   take((size_t)NN * DIM * 4);        // 16 MB
  float*    emb    = (float*)   take((size_t)QQ * DIM * 4);        // 2 MB
  float*    mean   = (float*)   take((size_t)R2 * DIM * 4);
  float*    invstd = (float*)   take((size_t)R2 * DIM * 4);
  int*      cnts   = (int*)     take((size_t)R2 * 4);
  int*      offs   = (int*)     take((size_t)(R2 + 1) * 4);
  int*      curs   = (int*)     take((size_t)R2 * 4);
  int*      perm   = (int*)     take((size_t)NN * 4);
  int*      pcnt   = (int*)     take((size_t)QQ * 4);
  int*      poff   = (int*)     take((size_t)(QQ + 1) * 4);
  int*      pcur   = (int*)     take((size_t)QQ * 4);
  int*      pperm  = (int*)     take((size_t)EE * 4);

  hipMemsetAsync(cnts, 0, (size_t)R2 * 4, stream);
  hipMemsetAsync(curs, 0, (size_t)R2 * 4, stream);
  hipMemsetAsync(pcnt, 0, (size_t)QQ * 4, stream);
  hipMemsetAsync(pcur, 0, (size_t)QQ * 4, stream);

  // 1) features -> f16
  compute_x_kernel<<<NN, 128, 0, stream>>>(neighbor_idx, years, months, days,
      ent_embs, y_freq, y_phi, y_amp, m_freq, m_phi, m_amp, d_freq, d_phi, d_amp, xh);

  // 2) W -> f16 (fits in L2: 15 MB)
  const int wtotal = R2 * DIM * DIM;
  convert_w_kernel<<<(wtotal + 255) / 256, 256, 0, stream>>>(W, Wh, wtotal);

  // 3) bucket neighbors by rel_id
  hist_kernel<<<(NN + 255) / 256, 256, 0, stream>>>(rel_id, cnts, NN);
  exscan_kernel<<<1, 1, 0, stream>>>(cnts, offs, R2);
  scatter_kernel<<<(NN + 255) / 256, 256, 0, stream>>>(rel_id, offs, curs, perm, NN);

  // 4) per-relation WMMA GEMM: z = x @ W[rel] + b[rel]
  gemm_per_rel_kernel<<<R2, 256, 0, stream>>>(xh, Wh, bvec, offs, perm, z);

  // 5) per-relation BatchNorm + ReLU
  bn_stats_kernel<<<R2, 128, 0, stream>>>(z, offs, perm, mean, invstd);
  bn_relu_kernel<<<NN, 128, 0, stream>>>(z, rel_id, offs, mean, invstd, gamma, beta);

  // 6) bucket pooling edges by dst, then average-pool
  hist_kernel<<<(EE + 255) / 256, 256, 0, stream>>>(pool_dst, pcnt, EE);
  exscan_kernel<<<1, 1, 0, stream>>>(pcnt, poff, QQ);
  scatter_kernel<<<(EE + 255) / 256, 256, 0, stream>>>(pool_dst, poff, pcur, pperm, EE);
  pool_avg_kernel<<<QQ, 128, 0, stream>>>(z, pool_src, poff, pperm, emb);

  // 7) TransE scoring
  score_kernel<<<BB, 128, 0, stream>>>(emb, rel_embs, head_pos, tail_pos, rels, out);
}
